// BioGuardGAT_46428596469972
// MI455X (gfx1250) — compile-verified, matrix operand order checked by reading, not run on hardware
//
#include <hip/hip_runtime.h>
#include <hip/hip_bf16.h>

// ---------------------------------------------------------------------------
// Problem constants (from the reference)
// ---------------------------------------------------------------------------
#define NN      65536      // nodes
#define NPG     32         // nodes per graph
#define GG      2048       // graphs
#define EE      196608     // edges
#define DD      128        // hidden dim
#define HH      4          // heads (layer 1)
#define MAXE    384        // max edges (incl. 32 self loops) kept per graph

typedef __bf16 bf16_t;
typedef __bf16 v16bf __attribute__((ext_vector_type(16)));
typedef float  v8f   __attribute__((ext_vector_type(8)));

// ---------------------------------------------------------------------------
// bf16 helpers (round-to-nearest-even convert)
// ---------------------------------------------------------------------------
static __device__ __forceinline__ bf16_t f2bf(float f) {
  unsigned u = __builtin_bit_cast(unsigned, f);
  unsigned r = u + 0x7FFFu + ((u >> 16) & 1u);
  unsigned short h = (unsigned short)(r >> 16);
  return __builtin_bit_cast(bf16_t, h);
}
static __device__ __forceinline__ float bf2f(bf16_t b) {
  unsigned short h = __builtin_bit_cast(unsigned short, b);
  unsigned u = ((unsigned)h) << 16;
  return __builtin_bit_cast(float, u);
}

// 32-byte fragment = two 16-byte halves (forces global_load_b128 pairs)
struct frag32 { uint4 x, y; };

// ---------------------------------------------------------------------------
// fp32 -> bf16 pack (plain row-major) — not used for GEMM B anymore
// ---------------------------------------------------------------------------
__global__ void __launch_bounds__(256) pack_bf16_kernel(
    const float* __restrict__ src, bf16_t* __restrict__ dst, int n) {
  int i = blockIdx.x * 256 + threadIdx.x;
  if (i < n) dst[i] = f2bf(src[i]);
}

// ---------------------------------------------------------------------------
// fp32 -> bf16 pack into WMMA B-fragment order:
//   Bsw[((kTile*(Nout/16) + nt)*32 + lane)*16 + e] = B[kTile*32 + kq][nt*16 + lane%16]
// where kq = (e<8 ? lo+e : lo+8+e), lo = (lane<16 ? 0 : 8).
// Each lane's 16 bf16 fragment elements become one contiguous 32B run.
// ---------------------------------------------------------------------------
__global__ void __launch_bounds__(256) pack_bsw_kernel(
    const float* __restrict__ B, bf16_t* __restrict__ Bsw, int K, int Nout) {
  int idx = blockIdx.x * 256 + threadIdx.x;
  if (idx >= K * Nout) return;
  const int NT    = Nout >> 4;
  const int e     = idx & 15;
  const int lane  = (idx >> 4) & 31;
  const int rest  = idx >> 9;
  const int nt    = rest % NT;
  const int kTile = rest / NT;
  const int lo    = (lane < 16) ? 0 : 8;
  const int kq    = (e < 8) ? (lo + e) : (lo + 8 + e);
  const int krow  = kTile * 32 + kq;
  const int ncol  = nt * 16 + (lane & 15);
  Bsw[idx] = f2bf(B[(size_t)krow * Nout + ncol]);
}

// ---------------------------------------------------------------------------
// Per-graph edge CSR build (graph id = dst_node >> 5)
// ---------------------------------------------------------------------------
__global__ void __launch_bounds__(256) edge_count_kernel(
    const int* __restrict__ edst, int* __restrict__ cnt) {
  int i = blockIdx.x * 256 + threadIdx.x;
  if (i < EE) atomicAdd(&cnt[edst[i] >> 5], 1);
}
__global__ void edge_scan_kernel(const int* __restrict__ cnt,
                                 int* __restrict__ ofs, int* __restrict__ cur) {
  if (threadIdx.x == 0 && blockIdx.x == 0) {
    int run = 0;
    for (int g = 0; g < GG; ++g) { ofs[g] = run; cur[g] = run; run += cnt[g]; }
    ofs[GG] = run;
  }
}
__global__ void __launch_bounds__(256) edge_scatter_kernel(
    const int* __restrict__ edst, int* __restrict__ cur, int* __restrict__ eord) {
  int i = blockIdx.x * 256 + threadIdx.x;
  if (i < EE) { int p = atomicAdd(&cur[edst[i] >> 5], 1); eord[p] = i; }
}

// ---------------------------------------------------------------------------
// Encoder: h0 = relu(bn(x @ enc_w + enc_b))  -> bf16 [N,128]
// ---------------------------------------------------------------------------
__global__ void __launch_bounds__(128) encoder_kernel(
    const float* __restrict__ x, const float* __restrict__ w,
    const float* __restrict__ b, const float* __restrict__ gam,
    const float* __restrict__ bet, bf16_t* __restrict__ h0) {
  __shared__ float xs[41];
  const int n = blockIdx.x, t = threadIdx.x;
  if (t < 41) xs[t] = x[(size_t)n * 41 + t];
  __syncthreads();
  float acc = b[t];
  #pragma unroll
  for (int k = 0; k < 41; ++k) acc += xs[k] * w[k * 128 + t];
  acc = acc * (gam[t] * rsqrtf(1.00001f)) + bet[t];
  acc = fmaxf(acc, 0.0f);
  h0[(size_t)n * 128 + t] = f2bf(acc);
}

// ---------------------------------------------------------------------------
// bf16 WMMA GEMM:  C[M,Nout](bf16) = A[M,K](bf16,row-major) @ Bsw + bias
// No LDS, no barriers. 256 threads = 8 waves; each wave owns a 32x32 output
// tile (4 wmma accumulators). A fragments = two global_load_b128 straight
// from the row-major activation (runs are contiguous & 16B aligned); B
// fragments = two global_load_b128 from the pre-swizzled weights. 8 waves of
// a block share one B tile (nTile = blockIdx.y), so B stays L1-hot.
// ---------------------------------------------------------------------------
__global__ void __launch_bounds__(256) gemm_bf16_wmma_kernel(
    const bf16_t* __restrict__ A, const bf16_t* __restrict__ Bsw,
    const float* __restrict__ bias, bf16_t* __restrict__ C,
    int M, int K, int Nout) {
  const int lane  = threadIdx.x & 31;
  const int wv    = threadIdx.x >> 5;
  const int mTile = blockIdx.x * 8 + wv;   // 32-row tile index
  const int nTile = blockIdx.y;            // 32-col tile index
  const int row0  = mTile * 32;
  const int NT    = Nout >> 4;

  const int mrow0 = row0 + (lane & 15);    // rows for mi=0 fragment
  const int mrow1 = mrow0 + 16;            // rows for mi=1 fragment
  const int lo    = (lane < 16) ? 0 : 8;

  v8f acc00 = {}, acc01 = {}, acc10 = {}, acc11 = {};

  for (int kk = 0; kk < K; kk += 32) {
    const int kTile = kk >> 5;
    // ---- A fragments: two contiguous 16B runs per row ----
    const bf16_t* a0p = A + (size_t)mrow0 * K + kk + lo;
    const bf16_t* a1p = A + (size_t)mrow1 * K + kk + lo;
    frag32 fa0, fa1;
    fa0.x = *(const uint4*)(a0p);
    fa0.y = *(const uint4*)(a0p + 16);
    fa1.x = *(const uint4*)(a1p);
    fa1.y = *(const uint4*)(a1p + 16);
    // ---- B fragments: contiguous 32B per lane from swizzled weights ----
    const bf16_t* bq = Bsw + ((size_t)(kTile * NT + nTile * 2) * 32 + lane) * 16;
    frag32 fb0, fb1;
    fb0.x = *(const uint4*)(bq);
    fb0.y = *(const uint4*)(bq + 8);
    fb1.x = *(const uint4*)(bq + 512);       // next 16-col sub-tile (+32 lanes*16)
    fb1.y = *(const uint4*)(bq + 512 + 8);

    const v16bf a0 = __builtin_bit_cast(v16bf, fa0);
    const v16bf a1 = __builtin_bit_cast(v16bf, fa1);
    const v16bf b0 = __builtin_bit_cast(v16bf, fb0);
    const v16bf b1 = __builtin_bit_cast(v16bf, fb1);

    acc00 = __builtin_amdgcn_wmma_f32_16x16x32_bf16(false, a0, false, b0,
                                                    (short)0, acc00, false, false);
    acc01 = __builtin_amdgcn_wmma_f32_16x16x32_bf16(false, a0, false, b1,
                                                    (short)0, acc01, false, false);
    acc10 = __builtin_amdgcn_wmma_f32_16x16x32_bf16(false, a1, false, b0,
                                                    (short)0, acc10, false, false);
    acc11 = __builtin_amdgcn_wmma_f32_16x16x32_bf16(false, a1, false, b1,
                                                    (short)0, acc11, false, false);
  }

  // store per f32 16x16 C/D layout: VGPR r -> M = r + (lane<16 ? 0 : 8)
  const int ncol0  = nTile * 32 + (lane & 15);
  const int ncol1  = ncol0 + 16;
  const int mbase0 = row0 + ((lane >> 4) << 3);
  const int mbase1 = mbase0 + 16;
  const float bv0 = bias[ncol0];
  const float bv1 = bias[ncol1];
  #pragma unroll
  for (int r = 0; r < 8; ++r) {
    C[(size_t)(mbase0 + r) * Nout + ncol0] = f2bf(acc00[r] + bv0);
    C[(size_t)(mbase0 + r) * Nout + ncol1] = f2bf(acc01[r] + bv1);
    C[(size_t)(mbase1 + r) * Nout + ncol0] = f2bf(acc10[r] + bv0);
    C[(size_t)(mbase1 + r) * Nout + ncol1] = f2bf(acc11[r] + bv1);
  }
}

// ---------------------------------------------------------------------------
// GAT layer 1 (4 heads x 128) fused per graph: self-loop attrs, logits,
// softmax, aggregation, +bias, ELU -> h1 bf16 [N,512].  Dynamic LDS 121216 B.
// ---------------------------------------------------------------------------
#define G1_SMEM 121216
__global__ void __launch_bounds__(256) gat1_kernel(
    const bf16_t* __restrict__ xl, const bf16_t* __restrict__ xr,
    const int* __restrict__ esrc, const int* __restrict__ edst,
    const int* __restrict__ eord, const int* __restrict__ ofs,
    const float* __restrict__ ea, const float* __restrict__ we,
    const float* __restrict__ att, const float* __restrict__ bias,
    bf16_t* __restrict__ h1) {
  extern __shared__ char smem[];
  bf16_t* xl_s   = (bf16_t*)(smem + 0);        // 32*512 bf16
  bf16_t* xr_s   = (bf16_t*)(smem + 32768);    // 32*512 bf16
  float*  wes    = (float*)(smem + 65536);     // 10*512
  float*  atts   = (float*)(smem + 86016);     // 512
  int*    els    = (int*)(smem + 88064);       // MAXE
  int*    eld    = (int*)(smem + 89600);       // MAXE
  float*  eattr  = (float*)(smem + 91136);     // MAXE*10
  float*  elog   = (float*)(smem + 106496);    // MAXE*4
  float*  ealpha = (float*)(smem + 112640);    // MAXE*4
  float*  deg    = (float*)(smem + 118784);    // 32
  float*  easum  = (float*)(smem + 118912);    // 32*10
  float*  mxs    = (float*)(smem + 120192);    // 128
  float*  invs   = (float*)(smem + 120704);    // 128

  const int tid = threadIdx.x;
  const int g   = blockIdx.x;
  const int n0  = g * NPG;

  {  // contiguous 32 KB node tiles -> LDS
    const uint4* s0 = (const uint4*)(xl + (size_t)n0 * 512);
    const uint4* s1 = (const uint4*)(xr + (size_t)n0 * 512);
    uint4* d0 = (uint4*)xl_s; uint4* d1 = (uint4*)xr_s;
    for (int p = tid; p < 2048; p += 256) { d0[p] = s0[p]; d1[p] = s1[p]; }
  }
  for (int p = tid; p < 5120; p += 256) wes[p] = we[p];
  for (int p = tid; p < 512;  p += 256) atts[p] = att[p];
  for (int p = tid; p < 32;   p += 256) deg[p] = 0.0f;
  for (int p = tid; p < 320;  p += 256) easum[p] = 0.0f;
  __syncthreads();

  const int base = ofs[g];
  int mraw = ofs[g + 1] - base;
  const int cnt = (mraw < (MAXE - NPG)) ? mraw : (MAXE - NPG);
  for (int i = tid; i < cnt; i += 256) {
    const int e = eord[base + i];
    const int s = esrc[e] - n0, d = edst[e] - n0;
    els[i] = s; eld[i] = d;
    #pragma unroll
    for (int k = 0; k < 10; ++k) {
      float v = ea[(size_t)e * 10 + k];
      eattr[i * 10 + k] = v;
      atomicAdd(&easum[d * 10 + k], v);
    }
    atomicAdd(&deg[d], 1.0f);
  }
  __syncthreads();
  if (tid < NPG) {  // self loops with mean incoming edge_attr
    const float dn = 1.0f / fmaxf(deg[tid], 1.0f);
    const int i = cnt + tid;
    els[i] = tid; eld[i] = tid;
    #pragma unroll
    for (int k = 0; k < 10; ++k) eattr[i * 10 + k] = easum[tid * 10 + k] * dn;
  }
  __syncthreads();
  const int ecnt = cnt + NPG;

  // logits per (edge, head); edge projection computed on the fly (K=10)
  for (int p = tid; p < ecnt * 4; p += 256) {
    const int i = p >> 2, h = p & 3, hb = h << 7;
    const bf16_t* xls = xl_s + els[i] * 512 + hb;
    const bf16_t* xrd = xr_s + eld[i] * 512 + hb;
    const float*  eat = eattr + i * 10;
    float acc = 0.0f;
    for (int c = 0; c < 128; ++c) {
      float eev = 0.0f;
      #pragma unroll
      for (int k = 0; k < 10; ++k) eev += eat[k] * wes[k * 512 + hb + c];
      float v = bf2f(xls[c]) + bf2f(xrd[c]) + eev;
      v = (v > 0.0f) ? v : 0.2f * v;
      acc += atts[hb + c] * v;
    }
    elog[p] = acc;
  }
  __syncthreads();

  if (tid < 128) {  // per (dst,head) softmax stats
    const int d = tid >> 2, h = tid & 3;
    float mx = -3.4e38f;
    for (int i = 0; i < ecnt; ++i)
      if (eld[i] == d) mx = fmaxf(mx, elog[i * 4 + h]);
    float s = 0.0f;
    for (int i = 0; i < ecnt; ++i)
      if (eld[i] == d) s += __expf(elog[i * 4 + h] - mx);
    mxs[tid] = mx; invs[tid] = 1.0f / (s + 1e-16f);
  }
  __syncthreads();

  for (int p = tid; p < ecnt * 4; p += 256) {
    const int i = p >> 2, h = p & 3, d = eld[i];
    ealpha[p] = __expf(elog[p] - mxs[d * 4 + h]) * invs[d * 4 + h];
  }
  __syncthreads();

  // aggregation: each wave owns dst nodes {w, w+8, w+16, w+24}; lanes = chans
  const int lane = tid & 31, wv = tid >> 5;
  for (int dd = 0; dd < 4; ++dd) {
    const int d = wv + (dd << 3);
    float acc[16];
    #pragma unroll
    for (int r = 0; r < 16; ++r) acc[r] = 0.0f;
    for (int i = 0; i < ecnt; ++i) {
      if (eld[i] != d) continue;                 // wave-uniform branch
      const bf16_t* xs = xl_s + els[i] * 512;
      const float*  al = ealpha + i * 4;
      #pragma unroll
      for (int r = 0; r < 16; ++r)               // head = r>>2 for c=lane+32r
        acc[r] += al[r >> 2] * bf2f(xs[lane + (r << 5)]);
    }
    bf16_t* dst = h1 + (size_t)(n0 + d) * 512;
    #pragma unroll
    for (int r = 0; r < 16; ++r) {
      const int c = lane + (r << 5);
      float v = acc[r] + bias[c];
      v = (v > 0.0f) ? v : (__expf(v) - 1.0f);   // ELU
      dst[c] = f2bf(v);
    }
  }
}

// ---------------------------------------------------------------------------
// GAT layer 2 (1 head x 128) fused per graph + ELU + mean/max pool
// -> va [G,256].  Dynamic LDS 61568 B.
// ---------------------------------------------------------------------------
#define G2_SMEM 61568
__global__ void __launch_bounds__(256) gat2_kernel(
    const bf16_t* __restrict__ xl, const bf16_t* __restrict__ xr,
    const int* __restrict__ esrc, const int* __restrict__ edst,
    const int* __restrict__ eord, const int* __restrict__ ofs,
    const float* __restrict__ ea, const float* __restrict__ we,
    const float* __restrict__ att, const float* __restrict__ bias,
    float* __restrict__ vout) {
  extern __shared__ char smem[];
  bf16_t* xl_s   = (bf16_t*)(smem + 0);        // 32*128 bf16
  bf16_t* xr_s   = (bf16_t*)(smem + 8192);     // 32*128 bf16
  float*  wes    = (float*)(smem + 16384);     // 10*128
  float*  atts   = (float*)(smem + 21504);     // 128
  int*    els    = (int*)(smem + 22016);       // MAXE
  int*    eld    = (int*)(smem + 23552);       // MAXE
  float*  eattr  = (float*)(smem + 25088);     // MAXE*10
  float*  elog   = (float*)(smem + 40448);     // MAXE
  float*  ealpha = (float*)(smem + 41984);     // MAXE
  float*  deg    = (float*)(smem + 43520);     // 32
  float*  easum  = (float*)(smem + 43648);     // 32*10
  float*  mxs    = (float*)(smem + 44928);     // 32
  float*  invs   = (float*)(smem + 45056);     // 32
  float*  h2s    = (float*)(smem + 45184);     // 32*128

  const int tid = threadIdx.x;
  const int g   = blockIdx.x;
  const int n0  = g * NPG;

  {
    const uint4* s0 = (const uint4*)(xl + (size_t)n0 * 128);
    const uint4* s1 = (const uint4*)(xr + (size_t)n0 * 128);
    uint4* d0 = (uint4*)xl_s; uint4* d1 = (uint4*)xr_s;
    for (int p = tid; p < 512; p += 256) { d0[p] = s0[p]; d1[p] = s1[p]; }
  }
  for (int p = tid; p < 1280; p += 256) wes[p] = we[p];
  for (int p = tid; p < 128;  p += 256) atts[p] = att[p];
  for (int p = tid; p < 32;   p += 256) deg[p] = 0.0f;
  for (int p = tid; p < 320;  p += 256) easum[p] = 0.0f;
  __syncthreads();

  const int base = ofs[g];
  int mraw = ofs[g + 1] - base;
  const int cnt = (mraw < (MAXE - NPG)) ? mraw : (MAXE - NPG);
  for (int i = tid; i < cnt; i += 256) {
    const int e = eord[base + i];
    const int s = esrc[e] - n0, d = edst[e] - n0;
    els[i] = s; eld[i] = d;
    #pragma unroll
    for (int k = 0; k < 10; ++k) {
      float v = ea[(size_t)e * 10 + k];
      eattr[i * 10 + k] = v;
      atomicAdd(&easum[d * 10 + k], v);
    }
    atomicAdd(&deg[d], 1.0f);
  }
  __syncthreads();
  if (tid < NPG) {
    const float dn = 1.0f / fmaxf(deg[tid], 1.0f);
    const int i = cnt + tid;
    els[i] = tid; eld[i] = tid;
    #pragma unroll
    for (int k = 0; k < 10; ++k) eattr[i * 10 + k] = easum[tid * 10 + k] * dn;
  }
  __syncthreads();
  const int ecnt = cnt + NPG;

  for (int p = tid; p < ecnt; p += 256) {
    const bf16_t* xls = xl_s + els[p] * 128;
    const bf16_t* xrd = xr_s + eld[p] * 128;
    const float*  eat = eattr + p * 10;
    float acc = 0.0f;
    for (int c = 0; c < 128; ++c) {
      float eev = 0.0f;
      #pragma unroll
      for (int k = 0; k < 10; ++k) eev += eat[k] * wes[k * 128 + c];
      float v = bf2f(xls[c]) + bf2f(xrd[c]) + eev;
      v = (v > 0.0f) ? v : 0.2f * v;
      acc += atts[c] * v;
    }
    elog[p] = acc;
  }
  __syncthreads();

  if (tid < NPG) {
    const int d = tid;
    float mx = -3.4e38f;
    for (int i = 0; i < ecnt; ++i) if (eld[i] == d) mx = fmaxf(mx, elog[i]);
    float s = 0.0f;
    for (int i = 0; i < ecnt; ++i) if (eld[i] == d) s += __expf(elog[i] - mx);
    mxs[d] = mx; invs[d] = 1.0f / (s + 1e-16f);
  }
  __syncthreads();
  for (int p = tid; p < ecnt; p += 256) {
    const int d = eld[p];
    ealpha[p] = __expf(elog[p] - mxs[d]) * invs[d];
  }
  __syncthreads();

  const int lane = tid & 31, wv = tid >> 5;
  for (int dd = 0; dd < 4; ++dd) {
    const int d = wv + (dd << 3);
    float acc[4] = {0.0f, 0.0f, 0.0f, 0.0f};
    for (int i = 0; i < ecnt; ++i) {
      if (eld[i] != d) continue;
      const bf16_t* xs = xl_s + els[i] * 128;
      const float a = ealpha[i];
      #pragma unroll
      for (int r = 0; r < 4; ++r) acc[r] += a * bf2f(xs[lane + (r << 5)]);
    }
    #pragma unroll
    for (int r = 0; r < 4; ++r) {
      const int c = lane + (r << 5);
      float v = acc[r] + bias[c];
      v = (v > 0.0f) ? v : (__expf(v) - 1.0f);   // ELU
      h2s[d * 128 + c] = v;
    }
  }
  __syncthreads();

  if (tid < 128) {  // mean/max pool over 32 nodes
    const int c = tid;
    float s = 0.0f, mx = -3.4e38f;
    for (int n = 0; n < NPG; ++n) {
      float v = h2s[n * 128 + c];
      s += v; mx = fmaxf(mx, v);
    }
    vout[(size_t)g * 256 + c]       = s * (1.0f / NPG);
    vout[(size_t)g * 256 + 128 + c] = mx;
  }
}

// ---------------------------------------------------------------------------
// MLP head per graph: combined[768] -> fc1(relu(bn)) -> fc2(relu) -> out
// ---------------------------------------------------------------------------
__global__ void __launch_bounds__(256) mlp_kernel(
    const float* __restrict__ va, const float* __restrict__ vb,
    const float* __restrict__ fw1, const float* __restrict__ fb1,
    const float* __restrict__ g1, const float* __restrict__ b1,
    const float* __restrict__ fw2, const float* __restrict__ fb2,
    const float* __restrict__ ow, const float* __restrict__ ob,
    float* __restrict__ out) {
  __shared__ float comb[768], h1m[256], h2m[64];
  const int g = blockIdx.x, t = threadIdx.x;
  {
    float a = va[(size_t)g * 256 + t], b = vb[(size_t)g * 256 + t];
    comb[t] = a + b; comb[256 + t] = fabsf(a - b); comb[512 + t] = a * b;
  }
  __syncthreads();
  {
    float acc = fb1[t];
    for (int k = 0; k < 768; ++k) acc += comb[k] * fw1[k * 256 + t];
    acc = acc * (g1[t] * rsqrtf(1.00001f)) + b1[t];
    h1m[t] = fmaxf(acc, 0.0f);
  }
  __syncthreads();
  if (t < 64) {
    float acc = fb2[t];
    for (int k = 0; k < 256; ++k) acc += h1m[k] * fw2[k * 64 + t];
    h2m[t] = fmaxf(acc, 0.0f);
  }
  __syncthreads();
  if (t == 0) {
    float acc = ob[0];
    for (int k = 0; k < 64; ++k) acc += h2m[k] * ow[k];
    out[g] = acc;
  }
}

// ---------------------------------------------------------------------------
// Host orchestration
// ---------------------------------------------------------------------------
extern "C" void kernel_launch(void* const* d_in, const int* in_sizes, int n_in,
                              void* d_out, int out_size, void* d_ws, size_t ws_size,
                              hipStream_t stream) {
  (void)in_sizes; (void)n_in; (void)out_size; (void)ws_size;

  const float* xa     = (const float*)d_in[0];
  const float* eaa    = (const float*)d_in[1];
  const float* xb     = (const float*)d_in[2];
  const float* eab    = (const float*)d_in[3];
  const int*   eia    = (const int*)d_in[4];
  const int*   eib    = (const int*)d_in[5];
  // d_in[6], d_in[7] = batch arrays (implicit: graph = node >> 5)
  const float* enc_w  = (const float*)d_in[8];
  const float* enc_b  = (const float*)d_in[9];
  const float* enc_g  = (const float*)d_in[10];
  const float* enc_bt = (const float*)d_in[11];
  const float* c1_wl  = (const float*)d_in[12];
  const float* c1_bl  = (const float*)d_in[13];
  const float* c1_wr  = (const float*)d_in[14];
  const float* c1_br  = (const float*)d_in[15];
  const float* c1_we  = (const float*)d_in[16];
  const float* c1_att = (const float*)d_in[17];
  const float* c1_bias= (const float*)d_in[18];
  const float* c2_wl  = (const float*)d_in[19];
  const float* c2_bl  = (const float*)d_in[20];
  const float* c2_wr  = (const float*)d_in[21];
  const float* c2_br  = (const float*)d_in[22];
  const float* c2_we  = (const float*)d_in[23];
  const float* c2_att = (const float*)d_in[24];
  const float* c2_bias= (const float*)d_in[25];
  const float* fc1_w  = (const float*)d_in[26];
  const float* fc1_b  = (const float*)d_in[27];
  const float* bn1_g  = (const float*)d_in[28];
  const float* bn1_b  = (const float*)d_in[29];
  const float* fc2_w  = (const float*)d_in[30];
  const float* fc2_b  = (const float*)d_in[31];
  const float* out_w  = (const float*)d_in[32];
  const float* out_b  = (const float*)d_in[33];

  // ---- workspace carve (256B aligned regions) ----
  char* W = (char*)d_ws;
  size_t off = 0;
  auto alloc = [&](size_t bytes) -> void* {
    void* p = W + off;
    off += (bytes + 255) & ~(size_t)255;
    return p;
  };
  bf16_t* h0  = (bf16_t*)alloc((size_t)NN * 128 * 2);
  bf16_t* xl1 = (bf16_t*)alloc((size_t)NN * 512 * 2);
  bf16_t* xr1 = (bf16_t*)alloc((size_t)NN * 512 * 2);
  bf16_t* h1  = (bf16_t*)alloc((size_t)NN * 512 * 2);
  bf16_t* xl2 = (bf16_t*)alloc((size_t)NN * 128 * 2);
  bf16_t* xr2 = (bf16_t*)alloc((size_t)NN * 128 * 2);
  float*  va  = (float*)alloc((size_t)GG * 256 * 4);
  float*  vb  = (float*)alloc((size_t)GG * 256 * 4);
  bf16_t* w1l = (bf16_t*)alloc(128 * 512 * 2);   // swizzled B fragments
  bf16_t* w1r = (bf16_t*)alloc(128 * 512 * 2);
  bf16_t* w2l = (bf16_t*)alloc(512 * 128 * 2);
  bf16_t* w2r = (bf16_t*)alloc(512 * 128 * 2);
  int* cnt  = (int*)alloc(GG * 4);
  int* ofs  = (int*)alloc((GG + 1) * 4);
  int* cur  = (int*)alloc(GG * 4);
  int* eord = (int*)alloc((size_t)EE * 4);

  // ---- pack projection weights into WMMA B-fragment order (once) ----
  pack_bsw_kernel<<<(128 * 512 + 255) / 256, 256, 0, stream>>>(c1_wl, w1l, 128, 512);
  pack_bsw_kernel<<<(128 * 512 + 255) / 256, 256, 0, stream>>>(c1_wr, w1r, 128, 512);
  pack_bsw_kernel<<<(512 * 128 + 255) / 256, 256, 0, stream>>>(c2_wl, w2l, 512, 128);
  pack_bsw_kernel<<<(512 * 128 + 255) / 256, 256, 0, stream>>>(c2_wr, w2r, 512, 128);

  for (int arm = 0; arm < 2; ++arm) {
    const float* x   = arm ? xb : xa;
    const float* ea  = arm ? eab : eaa;
    const int*   ei  = arm ? eib : eia;      // [2,E] flat: src = ei, dst = ei+E
    float*       vo  = arm ? vb : va;

    // per-graph edge CSR
    hipMemsetAsync(cnt, 0, GG * 4, stream);
    edge_count_kernel<<<EE / 256, 256, 0, stream>>>(ei + EE, cnt);
    edge_scan_kernel<<<1, 1, 0, stream>>>(cnt, ofs, cur);
    edge_scatter_kernel<<<EE / 256, 256, 0, stream>>>(ei + EE, cur, eord);

    // encoder
    encoder_kernel<<<NN, 128, 0, stream>>>(x, enc_w, enc_b, enc_g, enc_bt, h0);

    // GAT1 projections (WMMA): [N,128] @ [128,512], 32x32 tile per wave
    gemm_bf16_wmma_kernel<<<dim3(NN / 256, 512 / 32), 256, 0, stream>>>(
        h0, w1l, c1_bl, xl1, NN, 128, 512);
    gemm_bf16_wmma_kernel<<<dim3(NN / 256, 512 / 32), 256, 0, stream>>>(
        h0, w1r, c1_br, xr1, NN, 128, 512);

    // fused GAT1 attention + aggregation
    gat1_kernel<<<GG, 256, G1_SMEM, stream>>>(
        xl1, xr1, ei, ei + EE, eord, ofs, ea, c1_we, c1_att, c1_bias, h1);

    // GAT2 projections (WMMA): [N,512] @ [512,128]
    gemm_bf16_wmma_kernel<<<dim3(NN / 256, 128 / 32), 256, 0, stream>>>(
        h1, w2l, c2_bl, xl2, NN, 512, 128);
    gemm_bf16_wmma_kernel<<<dim3(NN / 256, 128 / 32), 256, 0, stream>>>(
        h1, w2r, c2_br, xr2, NN, 512, 128);

    // fused GAT2 attention + aggregation + ELU + mean/max pool
    gat2_kernel<<<GG, 256, G2_SMEM, stream>>>(
        xl2, xr2, ei, ei + EE, eord, ofs, ea, c2_we, c2_att, c2_bias, vo);
  }

  // head MLP
  mlp_kernel<<<GG, 256, 0, stream>>>(va, vb, fc1_w, fc1_b, bn1_g, bn1_b,
                                     fc2_w, fc2_b, out_w, out_b, (float*)d_out);
}
